// LoCo_48541720379659
// MI455X (gfx1250) — compile-verified, matrix operand order checked by reading, not run on hardware
//
#include <hip/hip_runtime.h>
#include <hip/hip_bf16.h>
#include <math.h>

typedef __attribute__((ext_vector_type(16))) __bf16 v16bf;
typedef __attribute__((ext_vector_type(8)))  __bf16 v8bf;
typedef __attribute__((ext_vector_type(8)))  float  v8f;

#define B_    32
#define N_    4096
#define D_    256
#define S_    24
#define MTOK  (B_ * N_)

__device__ __forceinline__ float sigmoidf_(float x) { return 1.f / (1.f + expf(-x)); }

// ---------------------------------------------------------------------------
// fp32 -> bf16 conversion (vectorized: float4 x2 in, 16B bf16 out)
// ---------------------------------------------------------------------------
__global__ __launch_bounds__(256)
void cvt_bf16_kernel(const float* __restrict__ in, __bf16* __restrict__ out, int n) {
    int i = (blockIdx.x * 256 + threadIdx.x) * 8;
    if (i + 7 < n) {
        float4 a = *(const float4*)&in[i];
        float4 b = *(const float4*)&in[i + 4];
        v8bf o;
        o[0] = (__bf16)a.x; o[1] = (__bf16)a.y; o[2] = (__bf16)a.z; o[3] = (__bf16)a.w;
        o[4] = (__bf16)b.x; o[5] = (__bf16)b.y; o[6] = (__bf16)b.z; o[7] = (__bf16)b.w;
        *(v8bf*)&out[i] = o;
    } else {
        for (int j = i; j < n; ++j) out[j] = (__bf16)in[j];
    }
}

// ---------------------------------------------------------------------------
// bf16 WMMA GEMM: C[M,N] = act(A[M,K] @ B[K,N] + bias)
// block tile 64x256, 8 waves (2x4), wave tile 32x64 = 2x4 v_wmma_f32_16x16x32_bf16
// A tile (64x32) staged via GLOBAL_LOAD_ASYNC_TO_LDS_B128 (ASYNCcnt path),
// B tile (32x256) staged transposed (Bt[n][k]) so both fragment types are two
// contiguous 16B ds_load_b128 per lane.
// ACT: 0=none, 1=relu.  OUT_BF16: 1 -> write Cb (bf16), 0 -> write Cf (fp32)
// M % 64 == 0, N % 256 == 0, K % 32 == 0  (all call sites satisfy this)
// ---------------------------------------------------------------------------
template<int ACT, int OUT_BF16>
__global__ __launch_bounds__(256)
void gemm_wmma_kernel(const __bf16* __restrict__ A, const __bf16* __restrict__ Bm,
                      const float* __restrict__ bias,
                      float* __restrict__ Cf, __bf16* __restrict__ Cb,
                      int M, int N, int K) {
    __shared__ __attribute__((aligned(16))) __bf16 As[64][40];   // [m][k], 80B pitch (16B-aligned rows)
    __shared__ __attribute__((aligned(16))) __bf16 Bt[256][40];  // [n][k], transposed tile

    const int tid  = threadIdx.x;
    const int lane = tid & 31;
    const int w    = tid >> 5;
    const int wm   = w >> 2;         // 0..1 (wave row, 32 rows each)
    const int wn   = w & 3;          // 0..3 (wave col, 64 cols each)
    const int half = lane >> 4;      // lane half-select
    const int idx  = lane & 15;      // m/n within fragment
    const int bm   = blockIdx.y * 64;
    const int bn   = blockIdx.x * 256;

    v8f zero = {};
    v8f acc[2][4];
    #pragma unroll
    for (int i = 0; i < 2; ++i)
        #pragma unroll
        for (int j = 0; j < 4; ++j) acc[i][j] = zero;

    // staging coordinates (loop-invariant)
    const int ar  = tid >> 2;              // A row 0..63
    const int ac  = (tid & 3) * 8;         // A col chunk 0/8/16/24
    const int bk0 = tid >> 5;              // B k row base for chunk0: 0..7
    const int bn0 = (tid & 31) * 8;        // B n chunk 0..248
    const unsigned lds_a = (unsigned)(size_t)&As[ar][ac];   // LDS byte offset (low 32 bits)

    for (int k0 = 0; k0 < K; k0 += 32) {
        // stage A tile 64x32: one async 16B memory->LDS DMA per lane (ASYNCcnt)
        {
            const __bf16* ga = &A[(size_t)(bm + ar) * K + (k0 + ac)];
            asm volatile("global_load_async_to_lds_b128 %0, %1, off"
                         :: "v"(lds_a), "v"(ga)
                         : "memory");
        }
        // stage B tile 32x256 transposed: four 16B vector loads per thread,
        // scattered as 8x b16 stores each into Bt[n][k]
        #pragma unroll
        for (int t = 0; t < 4; ++t) {
            int k  = bk0 + t * 8;
            v8bf v = *(const v8bf*)&Bm[(size_t)(k0 + k) * N + (bn + bn0)];
            #pragma unroll
            for (int e = 0; e < 8; ++e) Bt[bn0 + e][k] = v[e];
        }
        // hint next K-tile into cache (global_prefetch_b8)
        if (k0 + 32 < K) {
            __builtin_prefetch(&A[(size_t)(bm + ar) * K + (k0 + 32 + ac)], 0, 0);
            __builtin_prefetch(&Bm[(size_t)(k0 + 32 + bk0) * N + (bn + bn0)], 0, 0);
        }
        // async A DMA must complete before the barrier releases consumers
        asm volatile("s_wait_asynccnt 0x0" ::: "memory");
        __syncthreads();

        // fragments per CDNA5 16-bit layout:
        // lanes 0-15 hold K {0..7,16..23}; lanes 16-31 hold K {8..15,24..31}
        v16bf af[2], bfv[4];
        #pragma unroll
        for (int f = 0; f < 2; ++f) {
            int row = wm * 32 + f * 16 + idx;
            v8bf alo = *(const v8bf*)&As[row][half * 8];
            v8bf ahi = *(const v8bf*)&As[row][16 + half * 8];
            af[f] = __builtin_shufflevector(alo, ahi, 0, 1, 2, 3, 4, 5, 6, 7,
                                            8, 9, 10, 11, 12, 13, 14, 15);
        }
        #pragma unroll
        for (int f = 0; f < 4; ++f) {
            int col = wn * 64 + f * 16 + idx;
            v8bf blo = *(const v8bf*)&Bt[col][half * 8];
            v8bf bhi = *(const v8bf*)&Bt[col][16 + half * 8];
            bfv[f] = __builtin_shufflevector(blo, bhi, 0, 1, 2, 3, 4, 5, 6, 7,
                                             8, 9, 10, 11, 12, 13, 14, 15);
        }
        #pragma unroll
        for (int i = 0; i < 2; ++i)
            #pragma unroll
            for (int j = 0; j < 4; ++j)
                acc[i][j] = __builtin_amdgcn_wmma_f32_16x16x32_bf16(
                    false, af[i], false, bfv[j], (short)0, acc[i][j], false, false);
        __syncthreads();
    }

    // epilogue: C layout VGPR r -> row r + 8*half, col = idx (coalesced across lanes)
    #pragma unroll
    for (int i = 0; i < 2; ++i) {
        #pragma unroll
        for (int j = 0; j < 4; ++j) {
            int gn = bn + wn * 64 + j * 16 + idx;
            float bv = bias[gn];
            #pragma unroll
            for (int r = 0; r < 8; ++r) {
                int gm = bm + wm * 32 + i * 16 + half * 8 + r;
                float v = acc[i][j][r] + bv;
                if (ACT == 1) v = fmaxf(v, 0.f);
                if (OUT_BF16) Cb[(size_t)gm * N + gn] = (__bf16)v;
                else          Cf[(size_t)gm * N + gn] = v;
            }
        }
    }
}

// ---------------------------------------------------------------------------
// In-place LayerNorm over D=256, one wave32 per row
// ---------------------------------------------------------------------------
__global__ __launch_bounds__(256)
void ln_kernel(float* __restrict__ feat, const float* __restrict__ g,
               const float* __restrict__ beta, int rows) {
    int wrow = blockIdx.x * 8 + (threadIdx.x >> 5);
    if (wrow >= rows) return;
    int lane = threadIdx.x & 31;
    float* row = feat + (size_t)wrow * D_;
    float v[8], s = 0.f;
    #pragma unroll
    for (int j = 0; j < 8; ++j) { v[j] = row[lane + 32 * j]; s += v[j]; }
    #pragma unroll
    for (int o = 16; o; o >>= 1) s += __shfl_xor(s, o, 32);
    float mu = s * (1.f / 256.f);
    float var = 0.f;
    #pragma unroll
    for (int j = 0; j < 8; ++j) { float d = v[j] - mu; var += d * d; }
    #pragma unroll
    for (int o = 16; o; o >>= 1) var += __shfl_xor(var, o, 32);
    var *= (1.f / 256.f);
    float inv = rsqrtf(var + 1e-5f);
    #pragma unroll
    for (int j = 0; j < 8; ++j) {
        int d = lane + 32 * j;
        row[d] = (v[j] - mu) * inv * g[d] + beta[d];
    }
}

// ---------------------------------------------------------------------------
// pooled = mean_n feat; rho = sigmoid(relu(pooled@sp_w1+b1)@sp_w2+b2); one block/batch
// ---------------------------------------------------------------------------
__global__ __launch_bounds__(256)
void pooled_rho_kernel(const float* __restrict__ feat,
                       const float* __restrict__ sp_w1, const float* __restrict__ sp_b1,
                       const float* __restrict__ sp_w2, const float* __restrict__ sp_b2,
                       float* __restrict__ rho) {
    int b = blockIdx.x, d = threadIdx.x;
    __shared__ float pooled[256];
    __shared__ float red[256];
    float s = 0.f;
    for (int n = 0; n < N_; ++n) s += feat[((size_t)b * N_ + n) * D_ + d];
    pooled[d] = s * (1.f / (float)N_);
    __syncthreads();
    float h = sp_b1[d];
    for (int i = 0; i < 256; ++i) h += pooled[i] * sp_w1[i * 256 + d];
    h = fmaxf(h, 0.f);
    red[d] = h * sp_w2[d];   // sp_w2 is [256,1]
    __syncthreads();
    for (int st = 128; st > 0; st >>= 1) {
        if (d < st) red[d] += red[d + st];
        __syncthreads();
    }
    if (d == 0) rho[b] = sigmoidf_(red[0] + sp_b2[0]);
}

// ---------------------------------------------------------------------------
// slot prep: mu = repeat(centers,3), mu_sq, gh = mu@gru_whh + bhh; one block/slot
// ---------------------------------------------------------------------------
__global__ __launch_bounds__(256)
void slot_prep_kernel(const float* __restrict__ centers,
                      const float* __restrict__ gru_whh, const float* __restrict__ gru_bhh,
                      float* __restrict__ mu, float* __restrict__ mu_sq,
                      float* __restrict__ gh) {
    int s = blockIdx.x, t = threadIdx.x;
    int obj = s / 3;
    __shared__ float m[256];
    __shared__ float red[256];
    float v = centers[obj * 256 + t];
    m[t] = v;
    red[t] = v * v;
    mu[s * 256 + t] = v;
    __syncthreads();
    for (int st = 128; st > 0; st >>= 1) {
        if (t < st) red[t] += red[t + st];
        __syncthreads();
    }
    if (t == 0) mu_sq[s] = red[0];
    for (int c = t; c < 768; c += 256) {
        float a = gru_bhh[c];
        for (int i = 0; i < 256; ++i) a += m[i] * gru_whh[i * 768 + c];
        gh[s * 768 + c] = a;
    }
}

// ---------------------------------------------------------------------------
// attention: one wave32 per token, 8 tokens per wave; mu cached in LDS
// ---------------------------------------------------------------------------
__global__ __launch_bounds__(256)
void attn_kernel(const float* __restrict__ feat, const float* __restrict__ mu,
                 const float* __restrict__ mu_sq, const float* __restrict__ rho,
                 const float* __restrict__ level_times, float* __restrict__ attn) {
    __shared__ float smu[S_][256];
    __shared__ float smusq[S_];
    __shared__ float slt[3];
    int b = blockIdx.y;
    int tid = threadIdx.x, lane = tid & 31, w = tid >> 5;
    for (int i = tid; i < S_ * 256; i += 256) smu[i / 256][i % 256] = mu[i];
    if (tid < S_) smusq[tid] = mu_sq[tid];
    if (tid < 3) slt[tid] = level_times[tid];
    __syncthreads();
    float r = rho[b];

    for (int it = 0; it < 8; ++it) {
        int n = blockIdx.x * 64 + w * 8 + it;
        const float* f = feat + ((size_t)b * N_ + n) * D_;
        float fv[8];
        float acc[S_ + 1];
        #pragma unroll
        for (int k = 0; k <= S_; ++k) acc[k] = 0.f;
        #pragma unroll
        for (int j = 0; j < 8; ++j) {
            fv[j] = f[lane + 32 * j];
            acc[S_] += fv[j] * fv[j];
        }
        #pragma unroll
        for (int s = 0; s < S_; ++s)
            #pragma unroll
            for (int j = 0; j < 8; ++j) acc[s] += fv[j] * smu[s][lane + 32 * j];
        #pragma unroll
        for (int o = 16; o; o >>= 1)
            #pragma unroll
            for (int k = 0; k <= S_; ++k) acc[k] += __shfl_xor(acc[k], o, 32);

        float logit = -1e30f;
        if (lane < S_) {
            int lvl = lane % 3;
            float baseh = (lvl == 0) ? 1.f : ((lvl == 1) ? 2.f : 4.f);
            float hor = baseh * (1.f + r);
            float sq = acc[S_] + smusq[lane] - 2.f * acc[lane];
            float t = slt[lvl];
            float tau2 = t * t - sq;
            float dist = ((tau2 >= 0.f) ? 1.f : -1.f) * sqrtf(fabsf(tau2) + 1e-8f);
            float cone = tau2 / (hor * hor + 1e-8f);
            logit = -fabsf(dist) + 0.5f * tanhf(cone);   // TAU_TEMP = 1
        }
        float mx = logit;
        #pragma unroll
        for (int o = 16; o; o >>= 1) mx = fmaxf(mx, __shfl_xor(mx, o, 32));
        float e = (lane < S_) ? expf(logit - mx) : 0.f;
        float sum = e;
        #pragma unroll
        for (int o = 16; o; o >>= 1) sum += __shfl_xor(sum, o, 32);
        if (lane < S_) attn[((size_t)b * N_ + n) * S_ + lane] = e / sum;
    }
}

// ---------------------------------------------------------------------------
// agg[b,s,d] = sum_n attn[b,n,s] * feat[b,n,d]; one block per (s,b)
// ---------------------------------------------------------------------------
__global__ __launch_bounds__(256)
void agg_kernel(const float* __restrict__ attn, const float* __restrict__ feat,
                float* __restrict__ agg) {
    int s = blockIdx.x, b = blockIdx.y, d = threadIdx.x;
    float a = 0.f;
    for (int n = 0; n < N_; ++n) {
        float wv = attn[((size_t)b * N_ + n) * S_ + s];
        a += wv * feat[((size_t)b * N_ + n) * D_ + d];
    }
    agg[((size_t)b * S_ + s) * D_ + d] = a;
}

// ---------------------------------------------------------------------------
// GRU cell (torch r,z,n order), h = (1-z)*n + z*mu; one block per (b,s) slot
// ---------------------------------------------------------------------------
__global__ __launch_bounds__(256)
void gru_kernel(const float* __restrict__ agg, const float* __restrict__ gru_wih,
                const float* __restrict__ gru_bih, const float* __restrict__ gh,
                const float* __restrict__ mu, __bf16* __restrict__ hnb) {
    int bs = blockIdx.x;        // b*24 + s
    int s = bs % S_;
    int j = threadIdx.x;
    __shared__ float av[256];
    av[j] = agg[(size_t)bs * 256 + j];
    __syncthreads();
    float gr = gru_bih[j], gz = gru_bih[256 + j], gn = gru_bih[512 + j];
    for (int i = 0; i < 256; ++i) {
        float a = av[i];
        gr += a * gru_wih[i * 768 + j];
        gz += a * gru_wih[i * 768 + 256 + j];
        gn += a * gru_wih[i * 768 + 512 + j];
    }
    float rg = sigmoidf_(gr + gh[s * 768 + j]);
    float zg = sigmoidf_(gz + gh[s * 768 + 256 + j]);
    float ng = tanhf(gn + rg * gh[s * 768 + 512 + j]);
    float h = (1.f - zg) * ng + zg * mu[s * 256 + j];
    hnb[(size_t)bs * 256 + j] = (__bf16)h;
}

// ---------------------------------------------------------------------------
// recon[b,n,o] = sum_s attn[b,n,s] * slot_recon[b,s,o]; 16 tokens per block
// ---------------------------------------------------------------------------
__global__ __launch_bounds__(256)
void recon_kernel(const float* __restrict__ attn, const float* __restrict__ sr,
                  float* __restrict__ out) {
    int b = blockIdx.y;
    __shared__ float ssr[S_ * 256];
    __shared__ float sat[16][S_];
    for (int i = threadIdx.x; i < S_ * 256; i += 256) ssr[i] = sr[(size_t)b * S_ * 256 + i];
    int n0 = blockIdx.x * 16;
    for (int i = threadIdx.x; i < 16 * S_; i += 256)
        sat[i / S_][i % S_] = attn[((size_t)b * N_ + n0 + i / S_) * S_ + (i % S_)];
    __syncthreads();
    int o = threadIdx.x;
    for (int t = 0; t < 16; ++t) {
        float a = 0.f;
        #pragma unroll
        for (int s = 0; s < S_; ++s) a += sat[t][s] * ssr[s * 256 + o];
        out[((size_t)b * N_ + n0 + t) * D_ + o] = a;
    }
}

// ---------------------------------------------------------------------------
extern "C" void kernel_launch(void* const* d_in, const int* in_sizes, int n_in,
                              void* d_out, int out_size, void* d_ws, size_t ws_size,
                              hipStream_t stream) {
    const float* x        = (const float*)d_in[0];
    const float* enc_w1   = (const float*)d_in[1];
    const float* enc_b1   = (const float*)d_in[2];
    const float* enc_w2   = (const float*)d_in[3];
    const float* enc_b2   = (const float*)d_in[4];
    const float* enc_g    = (const float*)d_in[5];
    const float* enc_beta = (const float*)d_in[6];
    const float* sp_w1    = (const float*)d_in[7];
    const float* sp_b1    = (const float*)d_in[8];
    const float* sp_w2    = (const float*)d_in[9];
    const float* sp_b2    = (const float*)d_in[10];
    const float* centers  = (const float*)d_in[11];
    const float* lvl_t    = (const float*)d_in[12];
    const float* gru_wih  = (const float*)d_in[13];
    const float* gru_whh  = (const float*)d_in[14];
    const float* gru_bih  = (const float*)d_in[15];
    const float* gru_bhh  = (const float*)d_in[16];
    const float* dec_w1   = (const float*)d_in[17];
    const float* dec_b1   = (const float*)d_in[18];
    const float* dec_w2   = (const float*)d_in[19];
    const float* dec_b2   = (const float*)d_in[20];
    const float* dec_w3   = (const float*)d_in[21];
    const float* dec_b3   = (const float*)d_in[22];

    char* ws = (char*)d_ws;
    size_t off = 0;
    auto alloc = [&](size_t bytes) -> void* {
        void* p = ws + off;
        off += (bytes + 255) & ~(size_t)255;
        return p;
    };

    __bf16* xb   = (__bf16*)alloc((size_t)MTOK * 256 * 2);
    __bf16* w1b  = (__bf16*)alloc((size_t)256 * 512 * 2);
    __bf16* w2b  = (__bf16*)alloc((size_t)512 * 256 * 2);
    __bf16* dw1b = (__bf16*)alloc((size_t)256 * 512 * 2);
    __bf16* dw2b = (__bf16*)alloc((size_t)512 * 1024 * 2);
    __bf16* dw3b = (__bf16*)alloc((size_t)1024 * 256 * 2);
    __bf16* y1b  = (__bf16*)alloc((size_t)MTOK * 512 * 2);
    float*  feat = (float*)alloc((size_t)MTOK * 256 * 4);
    float*  rho  = (float*)alloc(32 * 4);
    float*  mu   = (float*)alloc(S_ * 256 * 4);
    float*  musq = (float*)alloc(S_ * 4);
    float*  gh   = (float*)alloc(S_ * 768 * 4);
    float*  attn = (float*)alloc((size_t)MTOK * S_ * 4);
    float*  agg  = (float*)alloc((size_t)B_ * S_ * 256 * 4);
    __bf16* hnb  = (__bf16*)alloc((size_t)B_ * S_ * 256 * 2);
    __bf16* d1b  = (__bf16*)alloc((size_t)B_ * S_ * 512 * 2);
    __bf16* d2b  = (__bf16*)alloc((size_t)B_ * S_ * 1024 * 2);
    float*  sr   = (float*)alloc((size_t)B_ * S_ * 256 * 4);

    auto cvt = [&](const float* src, __bf16* dst, int n) {
        cvt_bf16_kernel<<<(n / 8 + 255) / 256, 256, 0, stream>>>(src, dst, n);
    };
    cvt(x, xb, MTOK * 256);
    cvt(enc_w1, w1b, 256 * 512);
    cvt(enc_w2, w2b, 512 * 256);
    cvt(dec_w1, dw1b, 256 * 512);
    cvt(dec_w2, dw2b, 512 * 1024);
    cvt(dec_w3, dw3b, 1024 * 256);

    // encoder GEMM1 (relu, bf16 out) and GEMM2 (fp32 out), then in-place LN
    gemm_wmma_kernel<1, 1><<<dim3(512 / 256, MTOK / 64), 256, 0, stream>>>(
        xb, w1b, enc_b1, nullptr, y1b, MTOK, 512, 256);
    gemm_wmma_kernel<0, 0><<<dim3(256 / 256, MTOK / 64), 256, 0, stream>>>(
        y1b, w2b, enc_b2, feat, nullptr, MTOK, 256, 512);
    ln_kernel<<<MTOK / 8, 256, 0, stream>>>(feat, enc_g, enc_beta, MTOK);

    pooled_rho_kernel<<<B_, 256, 0, stream>>>(feat, sp_w1, sp_b1, sp_w2, sp_b2, rho);
    slot_prep_kernel<<<S_, 256, 0, stream>>>(centers, gru_whh, gru_bhh, mu, musq, gh);
    attn_kernel<<<dim3(N_ / 64, B_), 256, 0, stream>>>(feat, mu, musq, rho, lvl_t, attn);
    agg_kernel<<<dim3(S_, B_), 256, 0, stream>>>(attn, feat, agg);
    gru_kernel<<<B_ * S_, 256, 0, stream>>>(agg, gru_wih, gru_bih, gh, mu, hnb);

    // decoder MLP via WMMA (M = 768 rows)
    gemm_wmma_kernel<1, 1><<<dim3(512 / 256, (B_ * S_) / 64), 256, 0, stream>>>(
        hnb, dw1b, dec_b1, nullptr, d1b, B_ * S_, 512, 256);
    gemm_wmma_kernel<1, 1><<<dim3(1024 / 256, (B_ * S_) / 64), 256, 0, stream>>>(
        d1b, dw2b, dec_b2, nullptr, d2b, B_ * S_, 1024, 512);
    gemm_wmma_kernel<0, 0><<<dim3(256 / 256, (B_ * S_) / 64), 256, 0, stream>>>(
        d2b, dw3b, dec_b3, sr, nullptr, B_ * S_, 256, 1024);

    recon_kernel<<<dim3(N_ / 16, B_), 256, 0, stream>>>(attn, sr, (float*)d_out);
}
